// LinearMoE_44487271252124
// MI455X (gfx1250) — matrix operand order repaired and checked by reading, not compiled, not run on hardware
//
#include <hip/hip_runtime.h>
#include <hip/hip_bf16.h>
#include <stdint.h>

#define N_TOKENS 8192
#define D_IN     2048
#define HIDDEN   2048
#define N_EXPERTS 8
#define TOPK     2

typedef __attribute__((ext_vector_type(16))) __bf16 v16bf;
typedef __attribute__((ext_vector_type(8)))  __bf16 v8bf;
typedef __attribute__((ext_vector_type(8)))  float  v8f;

__device__ __forceinline__ unsigned short f2bf(float f) {
    unsigned u = __float_as_uint(f);
    u += 0x7FFFu + ((u >> 16) & 1u);   // round-to-nearest-even
    return (unsigned short)(u >> 16);
}

// ---------------------------------------------------------------------------
// Kernel 1: gating. One wave32 per token: logits = x@Wg + bg, softmax, top-2.
// ---------------------------------------------------------------------------
__global__ __launch_bounds__(256) void moe_gating_kernel(
    const float* __restrict__ x, const float* __restrict__ Wg,
    const float* __restrict__ bg,
    float* __restrict__ topw, int* __restrict__ topi)
{
    const int lane = threadIdx.x & 31;
    const int wave = threadIdx.x >> 5;
    const int t = blockIdx.x * 8 + wave;

    float a[N_EXPERTS];
#pragma unroll
    for (int e = 0; e < N_EXPERTS; ++e) a[e] = 0.f;

    const float* xrow = x + (size_t)t * D_IN;
    for (int d = lane; d < D_IN; d += 32) {
        const float xv = xrow[d];
        const float4* wr = (const float4*)(Wg + d * N_EXPERTS);
        const float4 w0 = wr[0], w1 = wr[1];
        a[0] += xv * w0.x; a[1] += xv * w0.y; a[2] += xv * w0.z; a[3] += xv * w0.w;
        a[4] += xv * w1.x; a[5] += xv * w1.y; a[6] += xv * w1.z; a[7] += xv * w1.w;
    }
#pragma unroll
    for (int e = 0; e < N_EXPERTS; ++e) {
#pragma unroll
        for (int off = 16; off > 0; off >>= 1)
            a[e] += __shfl_xor(a[e], off, 32);
    }
    if (lane == 0) {
        float l[N_EXPERTS];
        float m = -1e30f;
#pragma unroll
        for (int e = 0; e < N_EXPERTS; ++e) { l[e] = a[e] + bg[e]; m = fmaxf(m, l[e]); }
        float s = 0.f;
#pragma unroll
        for (int e = 0; e < N_EXPERTS; ++e) { l[e] = __expf(l[e] - m); s += l[e]; }
        const float inv = 1.f / s;
        int i0 = 0;
#pragma unroll
        for (int e = 1; e < N_EXPERTS; ++e) if (l[e] > l[i0]) i0 = e;   // ties -> lowest idx
        int i1 = (i0 == 0) ? 1 : 0;
#pragma unroll
        for (int e = 0; e < N_EXPERTS; ++e) if (e != i0 && l[e] > l[i1]) i1 = e;
        topw[2 * t + 0] = l[i0] * inv;
        topw[2 * t + 1] = l[i1] * inv;
        topi[2 * t + 0] = i0;
        topi[2 * t + 1] = i1;
    }
}

// ---------------------------------------------------------------------------
// Kernel 2: scatter (token,k) pairs into per-expert buckets.
// ---------------------------------------------------------------------------
__global__ void moe_scatter_kernel(const int* __restrict__ topi,
                                   int* __restrict__ cnt, int* __restrict__ bucket)
{
    const int pid = blockIdx.x * 256 + threadIdx.x;     // 0 .. 2*N_TOKENS-1
    const int e = topi[pid];
    const int slot = atomicAdd(&cnt[e], 1);
    bucket[e * N_TOKENS + slot] = pid;                  // token = pid>>1
}

// ---------------------------------------------------------------------------
// Kernel 3: convert x (f32) -> bf16, 8 elements/thread.
// ---------------------------------------------------------------------------
__global__ void moe_cvt_x_kernel(const float* __restrict__ x,
                                 unsigned short* __restrict__ xbf)
{
    const size_t i = ((size_t)blockIdx.x * 256 + threadIdx.x) * 8;
    const float4 p = *(const float4*)(x + i);
    const float4 q = *(const float4*)(x + i + 4);
    uint4 o;
    o.x = (unsigned)f2bf(p.x) | ((unsigned)f2bf(p.y) << 16);
    o.y = (unsigned)f2bf(p.z) | ((unsigned)f2bf(p.w) << 16);
    o.z = (unsigned)f2bf(q.x) | ((unsigned)f2bf(q.y) << 16);
    o.w = (unsigned)f2bf(q.z) | ((unsigned)f2bf(q.w) << 16);
    *(uint4*)(xbf + i) = o;
}

// ---------------------------------------------------------------------------
// Kernel 4: transpose-convert We [E][D][H] f32 -> WeT [E][H][D] bf16.
// ---------------------------------------------------------------------------
__global__ __launch_bounds__(256) void moe_cvt_we_kernel(
    const float* __restrict__ We, unsigned short* __restrict__ WeT)
{
    __shared__ float tile[32][33];
    const int e  = blockIdx.z;
    const int h0 = blockIdx.x * 32;
    const int d0 = blockIdx.y * 32;
    const int tx = threadIdx.x & 31;
    const int ty = threadIdx.x >> 5;                    // 0..7

    const float* src = We + ((size_t)e * D_IN + d0) * HIDDEN + h0;
#pragma unroll
    for (int i = 0; i < 32; i += 8)
        tile[ty + i][tx] = src[(size_t)(ty + i) * HIDDEN + tx];
    __syncthreads();
    unsigned short* dst = WeT + ((size_t)e * HIDDEN + h0) * D_IN + d0;
#pragma unroll
    for (int i = 0; i < 32; i += 8)
        dst[(size_t)(ty + i) * D_IN + tx] = f2bf(tile[tx][ty + i]);
}

// ---------------------------------------------------------------------------
// Kernel 5: grouped GEMM per expert bucket, bf16 WMMA, f32 accumulate.
// Block = 256 threads = 8 waves, tile 128(M) x 128(N), K-step 32.
// B tile is double-buffered in LDS and filled with the CDNA5 async path
// (global_load_async_to_lds_b128, ASYNCcnt), overlapping copy with WMMA.
// A fragments are register double-buffered.
// ---------------------------------------------------------------------------
#define BS_STRIDE 40   // ushorts per N-row of LDS B tile (32 data + 8 pad = 80B)

__global__ __launch_bounds__(256) void moe_gemm_kernel(
    const unsigned short* __restrict__ xbf,
    const unsigned short* __restrict__ WeT,
    const float* __restrict__ be,
    const int* __restrict__ bucket,
    const int* __restrict__ cnt,
    const float* __restrict__ topw,
    float* __restrict__ out)
{
    const int e = blockIdx.z;
    const int count = cnt[e];
    const int m0 = blockIdx.y * 128;
    if (m0 >= count) return;                            // over-provisioned grid
    const int n0 = blockIdx.x * 128;

    const int tid   = threadIdx.x;
    const int wave  = tid >> 5;
    const int lane  = tid & 31;
    const int laneM = lane & 15;
    const int kHalf = lane >> 4;

    __shared__ unsigned short Bs[2][128 * BS_STRIDE];

    // A operand: row laneM of this wave's 16-token strip (ISA A layout:
    // lanes 0-15 hold K 0-7 & 16-23, lanes 16-31 hold K 8-15 & 24-31).
    const int rowA = m0 + wave * 16 + laneM;
    const int rA   = (rowA < count) ? rowA : (count - 1);
    const int pidA = bucket[e * N_TOKENS + rA];
    const unsigned short* aBase = xbf + (size_t)(pidA >> 1) * D_IN + kHalf * 8;

    // Async B staging: thread -> (n row, K half), 32 bytes each stage.
    const int nL = tid >> 1, kh = tid & 1;
    const unsigned short* bGlob = WeT + ((size_t)e * HIDDEN + n0 + nL) * D_IN + kh * 16;
    const unsigned ldsOff = (unsigned)(nL * BS_STRIDE + kh * 16);
    const unsigned ldsA0 = (unsigned)(size_t)&Bs[0][ldsOff];   // low 32 bits = LDS offset
    const unsigned ldsA1 = (unsigned)(size_t)&Bs[1][ldsOff];

    // Prologue: stage k0=0 into buffer 0 (2 async b128 per thread).
    {
        const uint64_t ga = (uint64_t)bGlob;
        asm volatile(
            "global_load_async_to_lds_b128 %0, %1, off\n\t"
            "global_load_async_to_lds_b128 %0, %1, off offset:16"
            :: "v"(ldsA0), "v"(ga) : "memory");
    }

    v8f acc[8] = {};
    v8bf alo = *(const v8bf*)(aBase);
    v8bf ahi = *(const v8bf*)(aBase + 16);

    for (int k0 = 0; k0 < D_IN; k0 += 32) {
        const int cur = (k0 >> 5) & 1;
        // Last iteration re-stages k=0 into the dead buffer so the wait
        // immediate stays a constant 2 (pipeline remains branch-free).
        const int knext = (k0 + 32 < D_IN) ? (k0 + 32) : 0;

        // Issue next stage (async, into the other buffer).
        {
            const uint64_t ga = (uint64_t)(bGlob + knext);
            const unsigned ldsDst = cur ? ldsA0 : ldsA1;
            asm volatile(
                "global_load_async_to_lds_b128 %0, %1, off\n\t"
                "global_load_async_to_lds_b128 %0, %1, off offset:16"
                :: "v"(ldsDst), "v"(ga) : "memory");
        }
        // Register-prefetch next A fragment.
        const v8bf alo_n = *(const v8bf*)(aBase + knext);
        const v8bf ahi_n = *(const v8bf*)(aBase + knext + 16);

        // Own copies for the current buffer are done once only the 2
        // just-issued ops remain (async ops complete in order per wave).
        asm volatile("s_wait_asynccnt 2" ::: "memory");
        __syncthreads();   // all waves' copies for buffer `cur` visible

        const v16bf a = __builtin_shufflevector(alo, ahi,
            0,1,2,3,4,5,6,7,8,9,10,11,12,13,14,15);

#pragma unroll
        for (int j = 0; j < 8; ++j) {
            const unsigned short* bp = &Bs[cur][(j * 16 + laneM) * BS_STRIDE + kHalf * 8];
            const v8bf blo = *(const v8bf*)bp;
            const v8bf bhi = *(const v8bf*)(bp + 16);
            const v16bf b = __builtin_shufflevector(blo, bhi,
                0,1,2,3,4,5,6,7,8,9,10,11,12,13,14,15);
            acc[j] = __builtin_amdgcn_wmma_f32_16x16x32_bf16(
                false, a, false, b, (short)0, acc[j], false, false);
        }
        alo = alo_n; ahi = ahi_n;
        __syncthreads();   // buffer `cur` free for re-stage next iteration
    }

    // Epilogue: D element r -> row r + 8*kHalf, col laneM (+ tile offsets).
    const float* berow = be + (size_t)e * HIDDEN;
#pragma unroll
    for (int r = 0; r < 8; ++r) {
        const int row = m0 + wave * 16 + kHalf * 8 + r;
        if (row < count) {
            const int pid = bucket[e * N_TOKENS + row];
            const float w = topw[pid];
            float* orow = out + (size_t)(pid >> 1) * HIDDEN;
#pragma unroll
            for (int j = 0; j < 8; ++j) {
                const int col = n0 + j * 16 + laneM;
                unsafeAtomicAdd(&orow[col], w * (acc[j][r] + berow[col]));
            }
        }
    }
}

// ---------------------------------------------------------------------------
extern "C" void kernel_launch(void* const* d_in, const int* in_sizes, int n_in,
                              void* d_out, int out_size, void* d_ws, size_t ws_size,
                              hipStream_t stream)
{
    const float* x  = (const float*)d_in[0];
    const float* Wg = (const float*)d_in[1];
    const float* bg = (const float*)d_in[2];
    const float* We = (const float*)d_in[3];
    const float* be = (const float*)d_in[4];
    float* out = (float*)d_out;

    char* ws = (char*)d_ws;
    size_t off = 0;
    int* cnt = (int*)(ws + off);                 off += 256;
    float* topw = (float*)(ws + off);            off += (size_t)N_TOKENS * TOPK * sizeof(float);
    int* topi = (int*)(ws + off);                off += (size_t)N_TOKENS * TOPK * sizeof(int);
    int* bucket = (int*)(ws + off);              off += (size_t)N_EXPERTS * N_TOKENS * sizeof(int);
    unsigned short* xbf = (unsigned short*)(ws + off); off += (size_t)N_TOKENS * D_IN * 2;
    unsigned short* WeT = (unsigned short*)(ws + off); off += (size_t)N_EXPERTS * D_IN * HIDDEN * 2;

    hipMemsetAsync(cnt, 0, 256, stream);
    hipMemsetAsync(out, 0, (size_t)out_size * sizeof(float), stream);

    moe_gating_kernel<<<N_TOKENS / 8, 256, 0, stream>>>(x, Wg, bg, topw, topi);
    moe_scatter_kernel<<<(N_TOKENS * TOPK) / 256, 256, 0, stream>>>(topi, cnt, bucket);
    moe_cvt_x_kernel<<<(N_TOKENS * (size_t)D_IN) / (8 * 256), 256, 0, stream>>>(x, xbf);
    moe_cvt_we_kernel<<<dim3(HIDDEN / 32, D_IN / 32, N_EXPERTS), 256, 0, stream>>>(We, WeT);
    moe_gemm_kernel<<<dim3(HIDDEN / 128, N_TOKENS / 128, N_EXPERTS), 256, 0, stream>>>(
        xbf, WeT, be, bucket, cnt, topw, out);
}
